// JetBlock_8967891714369
// MI455X (gfx1250) — matrix operand order repaired and checked by reading, not compiled, not run on hardware
//
#include <hip/hip_runtime.h>
#include <hip/hip_bf16.h>
#include <stdint.h>

// ---------------------------------------------------------------------------
// JetBlock (gated delta rule) for MI455X / gfx1250 — wave32 + WMMA bf16.
// GEMM operands pre-converted to bf16 (B pre-transposed to [N][K]); GEMM hot
// loop: global_load_async_to_lds_b128 (ASYNCcnt) -> ds_load_b128 fragments ->
// v_wmma_f32_16x16x32_bf16, double-buffered, one barrier per K-step.
// ---------------------------------------------------------------------------

typedef __bf16 bf16x16 __attribute__((ext_vector_type(16)));
typedef float  f32x8   __attribute__((ext_vector_type(8)));

namespace {
constexpr int kB   = 2;
constexpr int kS   = 1024;
constexpr int kHID = 2048;
constexpr int kH   = 8;
constexpr int kDK  = 64;
constexpr int kDV  = 128;
constexpr int kTK  = kH * kDK;            // 512
constexpr int kTV  = kH * kDV;            // 1024
constexpr int kKC  = 4;                   // conv width
constexpr int kGH  = kTV / 8;             // 128
constexpr int kNQKV = kTK + kTK + kTV + kH + kH + kTV; // 3088
constexpr int kBS  = kB * kS;             // 2048 rows
constexpr float kEPS = 1e-6f;

constexpr int kOffQ = 0;
constexpr int kOffK = kTK;                // 512
constexpr int kOffV = 2 * kTK;            // 1024
constexpr int kOffA = 2 * kTK + kTV;      // 2048
constexpr int kOffB = kOffA + kH;         // 2056
constexpr int kOffZ = kOffB + kH;         // 2064
} // namespace

__device__ __forceinline__ float silu_f(float x)    { return x / (1.0f + __expf(-x)); }
__device__ __forceinline__ float sigmoid_f(float x) { return 1.0f / (1.0f + __expf(-x)); }
__device__ __forceinline__ float softplus_f(float x){ return (x > 20.0f) ? x : log1pf(__expf(x)); }
__device__ __forceinline__ unsigned short f2bf(float f) {
    unsigned u = __float_as_uint(f);
    u += 0x7FFFu + ((u >> 16) & 1u);      // round-to-nearest-even
    return (unsigned short)(u >> 16);
}
__device__ __forceinline__ unsigned lds_off(const void* p) {
    // ISA 10.2: for LDS apertures, the LDS address is the low 32 bits of the flat address
    return (unsigned)(unsigned long long)p;
}

// ---------------------------------------------------------------------------
// One-time f32 -> bf16 convert (contiguous).  n must be a multiple of 4.
// ---------------------------------------------------------------------------
__global__ __launch_bounds__(256) void cvt_bf16(
    const float* __restrict__ in, unsigned short* __restrict__ out, int n)
{
    int i = (blockIdx.x * 256 + threadIdx.x) * 4;
    if (i >= n) return;
    float4 v = *reinterpret_cast<const float4*>(in + i);
    ushort4 o;
    o.x = f2bf(v.x); o.y = f2bf(v.y); o.z = f2bf(v.z); o.w = f2bf(v.w);
    *reinterpret_cast<ushort4*>(out + i) = o;
}

// ---------------------------------------------------------------------------
// One-time f32 [K][N] -> bf16 [N][K] transpose-convert (LDS tiled 32x32).
// ---------------------------------------------------------------------------
__global__ __launch_bounds__(256) void cvt_bf16_t(
    const float* __restrict__ in, unsigned short* __restrict__ out, int K, int N)
{
    __shared__ unsigned short tile[32][33];
    int bk = blockIdx.y * 32, bn = blockIdx.x * 32;
    int tx = threadIdx.x & 31, ty = threadIdx.x >> 5;   // 32 x 8
    #pragma unroll
    for (int i = 0; i < 4; ++i) {
        int k = bk + ty + i * 8, n = bn + tx;
        float v = (k < K && n < N) ? in[(size_t)k * N + n] : 0.0f;
        tile[ty + i * 8][tx] = f2bf(v);
    }
    __syncthreads();
    #pragma unroll
    for (int i = 0; i < 4; ++i) {
        int n = bn + ty + i * 8, k = bk + tx;
        if (n < N && k < K) out[(size_t)n * K + k] = tile[tx][ty + i * 8];
    }
}

// ---------------------------------------------------------------------------
// WMMA bf16 GEMM:  C[M,N] = act( A[M,K] * Bt[N,K]^T + bias )
// Block tile 128x128, K-step 32, double-buffered LDS, one barrier per K-step.
// 8 waves; each wave: 32x64 subtile = 2x4 of 16x16 WMMA (8 v_wmma / K-step).
// GUARD_N=false path stages tiles with global_load_async_to_lds_b128
// (ASYNCcnt); GUARD_N=true keeps a register path so OOB rows are zero-filled.
// Assumes M % 128 == 0, K % 32 == 0, N % 8 == 0.
// ---------------------------------------------------------------------------
template <bool HAS_BIAS, bool ACT_SILU, bool OUT_BF16, bool GUARD_N>
__global__ __launch_bounds__(256) void wmma_gemm_bf16(
    const unsigned short* __restrict__ A, const unsigned short* __restrict__ Bt,
    void* __restrict__ Cout, const float* __restrict__ bias,
    int M, int N, int Kd)
{
    constexpr int TM = 128, TN = 128, TK = 32;
    __shared__ unsigned short As[2][TM][TK];   // 16 KB
    __shared__ unsigned short Bs[2][TN][TK];   // 16 KB ([n][k], fragments contiguous)

    const int tid  = threadIdx.x;
    const int lane = tid & 31;
    const int wave = tid >> 5;
    const int wm   = (wave & 3) * 32;          // wave row offset
    const int wn   = (wave >> 2) * 64;         // wave col offset
    const int bm   = blockIdx.y * TM;
    const int bn   = blockIdx.x * TN;
    const int nT   = Kd / TK;

    const int arow  = lane & 15;
    const int khalf = lane >> 4;

    uint4 ra[2], rb[2];                         // register path only

    // ---- staging helpers: 512 uint4 per 128x32 bf16 tile; 2 per thread ----
    auto stage_async = [&](int t, int buf) {    // GUARD_N == false
        int k0 = t * TK;
        #pragma unroll
        for (int i = 0; i < 2; ++i) {
            int u = tid + 256 * i;              // 0..511
            int row = u >> 2, j = u & 3;        // j = 16B chunk in 64B row
            unsigned la = lds_off(&As[buf][row][j * 8]);
            unsigned long long ga =
                (unsigned long long)(A + (size_t)(bm + row) * Kd + k0 + j * 8);
            asm volatile("global_load_async_to_lds_b128 %0, %1, off"
                         :: "v"(la), "v"(ga) : "memory");
            unsigned lb = lds_off(&Bs[buf][row][j * 8]);
            unsigned long long gb =
                (unsigned long long)(Bt + (size_t)(bn + row) * Kd + k0 + j * 8);
            asm volatile("global_load_async_to_lds_b128 %0, %1, off"
                         :: "v"(lb), "v"(gb) : "memory");
        }
    };
    auto load_regs = [&](int t) {               // GUARD_N == true
        int k0 = t * TK;
        #pragma unroll
        for (int i = 0; i < 2; ++i) {
            int u = tid + 256 * i;
            int row = u >> 2, j = u & 3;
            ra[i] = *reinterpret_cast<const uint4*>(
                A + (size_t)(bm + row) * Kd + k0 + j * 8);
            int gn  = bn + row;
            int gnc = gn < N ? gn : N - 1;      // branchless clamp
            uint4 v = *reinterpret_cast<const uint4*>(
                Bt + (size_t)gnc * Kd + k0 + j * 8);
            if (gn >= N) { v.x = 0u; v.y = 0u; v.z = 0u; v.w = 0u; }
            rb[i] = v;
        }
    };
    auto store_regs = [&](int buf) {
        #pragma unroll
        for (int i = 0; i < 2; ++i) {
            int u = tid + 256 * i;
            int row = u >> 2, j = u & 3;
            *reinterpret_cast<uint4*>(&As[buf][row][j * 8]) = ra[i];
            *reinterpret_cast<uint4*>(&Bs[buf][row][j * 8]) = rb[i];
        }
    };

    f32x8 acc[2][4] = {};

    // ---- prologue: stage tile 0 ----
    if constexpr (!GUARD_N) {
        stage_async(0, 0);
        asm volatile("s_wait_asynccnt 0x0" ::: "memory");
    } else {
        load_regs(0);
        store_regs(0);
    }
    __syncthreads();

    for (int t = 0; t < nT; ++t) {
        int buf = t & 1;

        if (t + 2 < nT) {                      // global_prefetch_b8 for t+2
            __builtin_prefetch(A  + (size_t)(bm + (tid >> 1)) * Kd + (t + 2) * TK, 0, 1);
            __builtin_prefetch(Bt + (size_t)(bn + (tid >> 1)) * Kd + (t + 2) * TK, 0, 1);
        }
        if (t + 1 < nT) {
            if constexpr (!GUARD_N) stage_async(t + 1, buf ^ 1);
            else                    load_regs(t + 1);
        }

        union Frag { bf16x16 v; uint4 q[2]; };
        Frag af[2], bfr[4];
        #pragma unroll
        for (int mt = 0; mt < 2; ++mt) {
            // 16-bit A 16x32 layout: lanes 0-15 K={0..7,16..23}; 16-31 K={8..15,24..31}
            const uint4* ap = reinterpret_cast<const uint4*>(&As[buf][wm + mt * 16 + arow][0]);
            af[mt].q[0] = ap[khalf];
            af[mt].q[1] = ap[2 + khalf];
        }
        #pragma unroll
        for (int nt = 0; nt < 4; ++nt) {
            // B 32x16 layout (matches A): lanes 0-15 K={0..7,16..23}; 16-31 K={8..15,24..31}
            const uint4* bp = reinterpret_cast<const uint4*>(&Bs[buf][wn + nt * 16 + arow][0]);
            bfr[nt].q[0] = bp[khalf];
            bfr[nt].q[1] = bp[2 + khalf];
        }
        #pragma unroll
        for (int mt = 0; mt < 2; ++mt)
            #pragma unroll
            for (int nt = 0; nt < 4; ++nt)
                acc[mt][nt] = __builtin_amdgcn_wmma_f32_16x16x32_bf16(
                    false, af[mt].v, false, bfr[nt].v,
                    (short)0, acc[mt][nt], false, false);

        if (t + 1 < nT) {
            if constexpr (!GUARD_N)
                asm volatile("s_wait_asynccnt 0x0" ::: "memory");
            else
                store_regs(buf ^ 1);
        }
        __syncthreads();
    }

    // ---- epilogue: row = r + khalf*8 in tile, col = lane&15; all branches
    //      resolved at compile time; bias loaded once per column ----
    #pragma unroll
    for (int nt = 0; nt < 4; ++nt) {
        const int col = bn + wn + nt * 16 + (lane & 15);
        bool ok = true;
        if constexpr (GUARD_N) ok = (col < N);
        float bv = 0.0f;
        if constexpr (HAS_BIAS) bv = ok ? bias[col] : 0.0f;
        #pragma unroll
        for (int mt = 0; mt < 2; ++mt) {
            const int row0 = bm + wm + mt * 16 + khalf * 8;
            #pragma unroll
            for (int r = 0; r < 8; ++r) {
                float v = acc[mt][nt][r];
                if constexpr (HAS_BIAS) v += bv;
                if constexpr (ACT_SILU) v = silu_f(v);
                if (ok) {
                    size_t idx = (size_t)(row0 + r) * N + col;
                    if constexpr (OUT_BF16)
                        ((unsigned short*)Cout)[idx] = f2bf(v);
                    else
                        ((float*)Cout)[idx] = v;
                }
            }
        }
    }
}

// ---------------------------------------------------------------------------
// q/k prep: SiLU -> L2 norm over DK=64 (q additionally scaled by DK^-1/2).
// ---------------------------------------------------------------------------
__global__ __launch_bounds__(256) void prep_qk(
    const float* __restrict__ qkvabz, float* __restrict__ qn, float* __restrict__ kn)
{
    int wave = blockIdx.x * 8 + (threadIdx.x >> 5);
    int lane = threadIdx.x & 31;
    int total = 2 * kBS * kH;
    if (wave >= total) return;
    int isk = wave >= kBS * kH;
    int r   = isk ? wave - kBS * kH : wave;
    int bs  = r >> 3, h = r & 7;
    const float* src = qkvabz + (size_t)bs * kNQKV + (isk ? kOffK : kOffQ) + h * kDK;
    float x0 = silu_f(src[2 * lane]);
    float x1 = silu_f(src[2 * lane + 1]);
    float ss = x0 * x0 + x1 * x1;
    #pragma unroll
    for (int m = 16; m >= 1; m >>= 1) ss += __shfl_xor(ss, m, 32);
    float rs = rsqrtf(ss + kEPS);
    if (!isk) rs *= 0.125f;                   // DK^-0.5
    float* dst = (isk ? kn : qn) + (size_t)r * kDK;
    dst[2 * lane]     = x0 * rs;
    dst[2 * lane + 1] = x1 * rs;
}

// ---------------------------------------------------------------------------
__global__ __launch_bounds__(256) void prep_gb(
    const float* __restrict__ qkvabz, const float* __restrict__ A_log,
    const float* __restrict__ dt_bias, float* __restrict__ g, float* __restrict__ beta)
{
    int i = blockIdx.x * 256 + threadIdx.x;
    if (i >= kBS * kH) return;
    int bs = i >> 3, h = i & 7;
    const float* row = qkvabz + (size_t)bs * kNQKV;
    g[i]    = -__expf(A_log[h]) * softplus_f(row[kOffA + h] + dt_bias[h]);
    beta[i] = sigmoid_f(row[kOffB + h]);
}

// ---------------------------------------------------------------------------
// Dynamic depthwise conv: v_conv[b,s,tv] = SiLU( sum_j xpad[b,s+j,tv]*kern[b,s,tv,j] )
// ---------------------------------------------------------------------------
__global__ __launch_bounds__(256) void dyn_conv(
    const float* __restrict__ qkvabz, const float* __restrict__ conv_state,
    const float* __restrict__ kern, float* __restrict__ vconv)
{
    size_t i = (size_t)blockIdx.x * 256 + threadIdx.x;
    if (i >= (size_t)kBS * kTV) return;
    int tv = (int)(i & (kTV - 1));
    size_t bs = i >> 10;
    int s = (int)(bs & (kS - 1));
    int b = (int)(bs >> 10);
    const float* kr = kern + i * kKC;
    float acc = 0.0f;
    #pragma unroll
    for (int j = 0; j < kKC; ++j) {
        int t = s + j;
        float x;
        if (t < kKC - 1)
            x = conv_state[((size_t)b * kTV + tv) * (kKC - 1) + t];
        else
            x = qkvabz[((size_t)(b * kS + (t - (kKC - 1)))) * kNQKV + kOffV + tv];
        acc = fmaf(x, kr[j], acc);
    }
    vconv[i] = silu_f(acc);
}

// ---------------------------------------------------------------------------
// Gated delta rule scan. One block per (b,h). 256 threads: thread = (v, khalf),
// khalf splits DK into two 32-halves (32 state regs/thread). Pairwise
// reductions via __shfl_xor(.,1) within the wave (wave32).
// ---------------------------------------------------------------------------
__global__ __launch_bounds__(256) void delta_scan(
    const float* __restrict__ qn, const float* __restrict__ kn,
    const float* __restrict__ vconv, const float* __restrict__ g,
    const float* __restrict__ beta, const float* __restrict__ init_state,
    float* __restrict__ obuf)
{
    int bh = blockIdx.x;
    int b = bh >> 3, h = bh & 7;
    int tid = threadIdx.x;
    int v = tid >> 1;
    int khalf = tid & 1;
    int kbase = khalf * 32;

    float st[32];
    const float* ist = init_state + (size_t)bh * kDK * kDV;
    #pragma unroll
    for (int i = 0; i < 32; ++i) st[i] = ist[(size_t)(kbase + i) * kDV + v];

    __shared__ float q_sh[kDK];
    __shared__ float k_sh[kDK];
    __shared__ float gb_sh[2];

    for (int s = 0; s < kS; ++s) {
        size_t r = (size_t)(b * kS + s) * kH + h;
        if (tid < kDK) {
            q_sh[tid] = qn[r * kDK + tid];
            k_sh[tid] = kn[r * kDK + tid];
        }
        if (tid == 0) { gb_sh[0] = g[r]; gb_sh[1] = beta[r]; }
        __syncthreads();

        float eg = __expf(gb_sh[0]);
        float kv = 0.0f;
        #pragma unroll
        for (int i = 0; i < 32; ++i) {
            st[i] *= eg;
            kv = fmaf(k_sh[kbase + i], st[i], kv);
        }
        kv += __shfl_xor(kv, 1, 32);

        float vt = vconv[((size_t)(b * kS + s)) * kTV + h * kDV + v];
        float delta = (vt - kv) * gb_sh[1];

        float o = 0.0f;
        #pragma unroll
        for (int i = 0; i < 32; ++i) {
            st[i] = fmaf(k_sh[kbase + i], delta, st[i]);
            o = fmaf(q_sh[kbase + i], st[i], o);
        }
        o += __shfl_xor(o, 1, 32);
        if (khalf == 0)
            obuf[((size_t)(b * kS + s)) * kTV + h * kDV + v] = o;
        __syncthreads();
    }
}

// ---------------------------------------------------------------------------
// Gated RMSNorm (gate first, then normalize); writes bf16 for the final GEMM.
// ---------------------------------------------------------------------------
__global__ __launch_bounds__(256) void gate_norm(
    const float* __restrict__ qkvabz, const float* __restrict__ w,
    const float* __restrict__ obuf, unsigned short* __restrict__ obf)
{
    int wave = blockIdx.x * 8 + (threadIdx.x >> 5);
    int lane = threadIdx.x & 31;
    if (wave >= kBS * kH) return;
    int bs = wave >> 3, h = wave & 7;
    const float* o = obuf + (size_t)bs * kTV + h * kDV;
    const float* z = qkvabz + (size_t)bs * kNQKV + kOffZ + h * kDV;
    float ov[4];
    float ss = 0.0f;
    #pragma unroll
    for (int e = 0; e < 4; ++e) {
        int c = e * 32 + lane;
        float val = o[c] * silu_f(z[c]);
        ov[e] = val;
        ss += val * val;
    }
    #pragma unroll
    for (int m = 16; m >= 1; m >>= 1) ss += __shfl_xor(ss, m, 32);
    float rs = rsqrtf(ss * (1.0f / kDV) + kEPS);
    unsigned short* dst = obf + (size_t)bs * kTV + h * kDV;
    #pragma unroll
    for (int e = 0; e < 4; ++e) {
        int c = e * 32 + lane;
        dst[c] = f2bf(ov[e] * rs * w[c]);
    }
}

// ---------------------------------------------------------------------------
extern "C" void kernel_launch(void* const* d_in, const int* in_sizes, int n_in,
                              void* d_out, int out_size, void* d_ws, size_t ws_size,
                              hipStream_t stream) {
    const float* hidden = (const float*)d_in[0];
    const float* Wqkv   = (const float*)d_in[1];
    const float* A_log  = (const float*)d_in[2];
    const float* dtb    = (const float*)d_in[3];
    const float* cstate = (const float*)d_in[4];
    const float* w1     = (const float*)d_in[5];
    const float* w2     = (const float*)d_in[6];
    const float* b2     = (const float*)d_in[7];
    const float* istate = (const float*)d_in[8];
    const float* onw    = (const float*)d_in[9];
    const float* Wo     = (const float*)d_in[10];
    float* out = (float*)d_out;

    // --- workspace carve-out (256B aligned) ---
    char* wp = (char*)d_ws;
    auto alloc = [&](size_t bytes) -> void* {
        void* r = (void*)wp;
        wp += (bytes + 255) & ~(size_t)255;
        return r;
    };
    float* qkvabz  = (float*)alloc((size_t)kBS * kNQKV * 4);
    float* kernbuf = (float*)alloc((size_t)kBS * kTV * kKC * 4);
    float* vconv   = (float*)alloc((size_t)kBS * kTV * 4);
    float* qn      = (float*)alloc((size_t)kBS * kTK * 4);
    float* kn      = (float*)alloc((size_t)kBS * kTK * 4);
    float* gbuf    = (float*)alloc((size_t)kBS * kH * 4);
    float* bbuf    = (float*)alloc((size_t)kBS * kH * 4);
    float* obuf    = (float*)alloc((size_t)kBS * kTV * 4);
    unsigned short* hid_bf = (unsigned short*)alloc((size_t)kBS * kHID * 2);
    unsigned short* Wqkv_t = (unsigned short*)alloc((size_t)kNQKV * kHID * 2);
    unsigned short* w1_t   = (unsigned short*)alloc((size_t)kGH * kHID * 2);
    unsigned short* w2_t   = (unsigned short*)alloc((size_t)(kTV * kKC) * kGH * 2);
    unsigned short* Wo_t   = (unsigned short*)alloc((size_t)kHID * kTV * 2);
    unsigned short* h1_bf  = (unsigned short*)alloc((size_t)kBS * kGH * 2);
    unsigned short* obf    = (unsigned short*)alloc((size_t)kBS * kTV * 2);

    dim3 blk(256);

    // --- one-time bf16 conversion / transposition of GEMM operands ---
    {
        int n = kBS * kHID;
        cvt_bf16<<<(n / 4 + 255) / 256, blk, 0, stream>>>(hidden, hid_bf, n);
    }
    cvt_bf16_t<<<dim3((kNQKV + 31) / 32, kHID / 32), blk, 0, stream>>>(Wqkv, Wqkv_t, kHID, kNQKV);
    cvt_bf16_t<<<dim3(kGH / 32, kHID / 32), blk, 0, stream>>>(w1, w1_t, kHID, kGH);
    cvt_bf16_t<<<dim3((kTV * kKC) / 32, kGH / 32), blk, 0, stream>>>(w2, w2_t, kGH, kTV * kKC);
    cvt_bf16_t<<<dim3(kHID / 32, kTV / 32), blk, 0, stream>>>(Wo, Wo_t, kTV, kHID);

    // 1) qkvabz = hidden @ W_qkvabz   (N=3088 ragged -> guarded register path)
    wmma_gemm_bf16<false, false, false, true>
        <<<dim3((kNQKV + 127) / 128, kBS / 128), blk, 0, stream>>>(
        hid_bf, Wqkv_t, qkvabz, nullptr, kBS, kNQKV, kHID);

    // 2) h1 = silu(hidden @ w1)       (bf16 out, async staging)
    wmma_gemm_bf16<false, true, true, false>
        <<<dim3(1, kBS / 128), blk, 0, stream>>>(
        hid_bf, w1_t, h1_bf, nullptr, kBS, kGH, kHID);

    // 3) kernels = h1 @ w2 + b2       (f32 out, async staging)
    wmma_gemm_bf16<true, false, false, false>
        <<<dim3((kTV * kKC) / 128, kBS / 128), blk, 0, stream>>>(
        h1_bf, w2_t, kernbuf, b2, kBS, kTV * kKC, kGH);

    // 4) q/k SiLU + L2 norm
    prep_qk<<<(2 * kBS * kH) / 8, blk, 0, stream>>>(qkvabz, qn, kn);

    // 5) g / beta
    prep_gb<<<(kBS * kH + 255) / 256, blk, 0, stream>>>(qkvabz, A_log, dtb, gbuf, bbuf);

    // 6) dynamic conv + SiLU
    dyn_conv<<<(unsigned)(((size_t)kBS * kTV + 255) / 256), blk, 0, stream>>>(
        qkvabz, cstate, kernbuf, vconv);

    // 7) recurrent gated delta rule
    delta_scan<<<kB * kH, blk, 0, stream>>>(qn, kn, vconv, gbuf, bbuf, istate, obuf);

    // 8) gated RMSNorm -> bf16
    gate_norm<<<(kBS * kH) / 8, blk, 0, stream>>>(qkvabz, onw, obuf, obf);

    // 9) out = o @ W_o                (f32 out, async staging)
    wmma_gemm_bf16<false, false, false, false>
        <<<dim3(kHID / 128, kBS / 128), blk, 0, stream>>>(
        obf, Wo_t, out, nullptr, kBS, kHID, kTV);
}
